// MultiHeadAttention_80187039416803
// MI455X (gfx1250) — matrix operand order repaired and checked by reading, not compiled, not run on hardware
//
#include <hip/hip_runtime.h>

// ---------------- problem constants ----------------
#define BATCH   2
#define SEQ     2048
#define NHEAD   16
#define DK      64
#define DMODEL  1024
#define SCALE   0.125f          // 1/sqrt(64)
#define KC      64              // key-chunk / k-chunk width

typedef __bf16 v8bf  __attribute__((ext_vector_type(8)));
typedef __bf16 v16bf __attribute__((ext_vector_type(16)));
typedef float  v8f   __attribute__((ext_vector_type(8)));

// bf16 WMMA: D(16x16,f32) = A(16x32,bf16) x B(32x16,bf16) + C
#define WMMA_BF16(a, b, c) \
  __builtin_amdgcn_wmma_f32_16x16x32_bf16(false, (a), false, (b), (short)0, (c), false, false)

// CDNA5 async memory->LDS DMA (per-lane b128), tracked by ASYNCcnt.
// ldsoff: 32-bit LDS byte offset VGPR; gptr: 64-bit global address VGPR pair.
#define ASYNC_LDS_B128(ldsoff, gptr)                                      \
  asm volatile("global_load_async_to_lds_b128 %0, %1, off"                \
               :: "v"((unsigned)(ldsoff)),                                \
                  "v"((unsigned long long)(uintptr_t)(gptr)) : "memory")

union ABHalves { v16bf v; v8bf h[2]; };

// ---------------- pre-pass: f32 -> bf16, head-major reshape ----------------
// q,k,v: [B][S][H*DK] f32.  Qb,Kb: [B*H][S][DK] bf16.  Vt: [B*H][DK][S] bf16.
__global__ void reshape_qkv_kernel(const float* __restrict__ q,
                                   const float* __restrict__ k,
                                   const float* __restrict__ v,
                                   __bf16* __restrict__ Qb,
                                   __bf16* __restrict__ Kb,
                                   __bf16* __restrict__ Vt) {
  int idx = blockIdx.x * 256 + threadIdx.x;            // over B*S*DMODEL
  if (idx >= BATCH * SEQ * DMODEL) return;
  int d = idx & (DK - 1);
  int h = (idx >> 6) & (NHEAD - 1);
  int s = (idx >> 10) & (SEQ - 1);
  int b = idx >> 21;                                    // / (SEQ*DMODEL)
  int bh = b * NHEAD + h;
  Qb[(bh * SEQ + s) * DK + d] = (__bf16)q[idx];
  Kb[(bh * SEQ + s) * DK + d] = (__bf16)k[idx];
  Vt[(bh * DK + d) * SEQ + s] = (__bf16)v[idx];
}

__global__ void convert_w_kernel(const float* __restrict__ w, __bf16* __restrict__ Wb) {
  int idx = blockIdx.x * 256 + threadIdx.x;
  if (idx >= DMODEL * DMODEL) return;
  Wb[idx] = (__bf16)w[idx];
}

// ---------------- flash attention ----------------
// grid: B*H * (S/128) blocks, 256 threads (8 waves, 16 q-rows each)
// LDS: K ping-pong 2x8KB | V^T ping-pong 2x8KB | per-wave P scratch 16KB
__global__ void __launch_bounds__(256)
flash_attn_kernel(const __bf16* __restrict__ Qb,
                  const __bf16* __restrict__ Kb,
                  const __bf16* __restrict__ Vt,
                  __bf16* __restrict__ ctx) {          // [B][S][DMODEL] bf16
  __shared__ __align__(16) char lds[2 * 8192 + 2 * 8192 + 8 * 2048];
  const unsigned ldsBase = (unsigned)(uintptr_t)(&lds[0]);
  const unsigned VOFF = 16384u, POFF = 32768u;

  const int bh   = blockIdx.x >> 4;                    // 0..31
  const int qblk = blockIdx.x & 15;                    // 0..15
  const int t    = threadIdx.x;
  const int wave = t >> 5;
  const int lane = t & 31;
  const int lh   = lane & 15;
  const int hi   = lane >> 4;                          // 0: lanes 0-15, 1: 16-31
  const int koff = hi ? 16 : 0;
  const int hi8  = hi ? 8 : 0;
  const int q0   = qblk * 128 + wave * 16;

  const __bf16* Kbh  = Kb + bh * (SEQ * DK);
  const __bf16* Vtbh = Vt + bh * (DK * SEQ);
  __bf16* Pw = (__bf16*)(lds + POFF + wave * 2048);    // [16][KC]

  // Q tile (16x64) in A-layout registers
  const __bf16* qrow = Qb + (bh * SEQ + q0 + lh) * DK;
  ABHalves a0, a1;
  a0.h[0] = *(const v8bf*)(qrow + koff);
  a0.h[1] = *(const v8bf*)(qrow + koff + 8);
  a1.h[0] = *(const v8bf*)(qrow + 32 + koff);
  a1.h[1] = *(const v8bf*)(qrow + 32 + koff + 8);

  v8f o[4] = {{}, {}, {}, {}};
  float mrow[8], lrow[8];
#pragma unroll
  for (int r = 0; r < 8; ++r) { mrow[r] = -INFINITY; lrow[r] = 0.0f; }

  // per-thread staging geometry (piece = 16B)
  const int vrow = t >> 3, vpart = t & 7;              // V pieces t and t+256

  // issue 4 async b128 loads (K x2, V x2) for chunk `c` into buffer `buf`
  auto stage = [&](int buf, int c) {
    const int k0 = c * KC;
    const unsigned kd = ldsBase + buf * 8192 + t * 16;
    const __bf16* ks = Kbh + k0 * DK + t * 8;          // contiguous 8KB chunk
    ASYNC_LDS_B128(kd,          ks);
    ASYNC_LDS_B128(kd + 4096u,  ks + 2048);
    const unsigned vd = ldsBase + VOFF + buf * 8192 + t * 16;
    const __bf16* vs = Vtbh + vrow * SEQ + k0 + vpart * 8;
    ASYNC_LDS_B128(vd,          vs);
    ASYNC_LDS_B128(vd + 4096u,  vs + 32 * SEQ);
  };

  stage(0, 0);                                         // prologue
  const int NCH = SEQ / KC;                            // 32

  for (int i = 0; i < NCH; ++i) {
    const int cur = i & 1;
    if (i + 1 < NCH) {
      stage(cur ^ 1, i + 1);                           // prefetch next chunk
      asm volatile("s_wait_asynccnt 0x4" ::: "memory");// chunk i landed (in-order)
    } else {
      asm volatile("s_wait_asynccnt 0x0" ::: "memory");
    }
    __syncthreads();                                   // visible to all waves

    const __bf16* Kl = (const __bf16*)(lds + cur * 8192);         // [key][dk]
    const __bf16* Vl = (const __bf16*)(lds + VOFF + cur * 8192);  // [dk][key]

    // ---- scores: S(16x64) = Q(16x64) . K^T, four 16-key tiles ----
    v8f s[4];
#pragma unroll
    for (int nk = 0; nk < 4; ++nk) {
      const __bf16* kp = Kl + (nk * 16 + lh) * DK + (hi ? 16 : 0);
      v8f acc = {};
      acc = WMMA_BF16(a0.v, *(const v16bf*)kp,        acc);
      acc = WMMA_BF16(a1.v, *(const v16bf*)(kp + 32), acc);
      s[nk] = acc;
    }

    // ---- online softmax (rows r+hi8; N striped across 16-lane halves) ----
#pragma unroll
    for (int r = 0; r < 8; ++r) {
      float x0 = s[0][r] * SCALE, x1 = s[1][r] * SCALE;
      float x2 = s[2][r] * SCALE, x3 = s[3][r] * SCALE;
      float m = fmaxf(fmaxf(x0, x1), fmaxf(x2, x3));
#pragma unroll
      for (int off = 8; off >= 1; off >>= 1)
        m = fmaxf(m, __shfl_xor(m, off, 32));
      float mnew = fmaxf(mrow[r], m);
      float corr = __expf(mrow[r] - mnew);
      float p0 = __expf(x0 - mnew), p1 = __expf(x1 - mnew);
      float p2 = __expf(x2 - mnew), p3 = __expf(x3 - mnew);
      float rs = (p0 + p1) + (p2 + p3);
#pragma unroll
      for (int off = 8; off >= 1; off >>= 1)
        rs += __shfl_xor(rs, off, 32);
      lrow[r] = lrow[r] * corr + rs;
      mrow[r] = mnew;
      o[0][r] *= corr; o[1][r] *= corr; o[2][r] *= corr; o[3][r] *= corr;
      __bf16* pr = Pw + (r + hi8) * KC + lh;
      pr[0]  = (__bf16)p0;  pr[16] = (__bf16)p1;
      pr[32] = (__bf16)p2;  pr[48] = (__bf16)p3;
    }

    // D-layout -> A-layout round trip through LDS (same wave: wait on DS only)
    asm volatile("s_wait_dscnt 0" ::: "memory");

    ABHalves pa0, pa1;
    pa0.h[0] = *(const v8bf*)(Pw + lh * KC + koff);
    pa0.h[1] = *(const v8bf*)(Pw + lh * KC + koff + 8);
    pa1.h[0] = *(const v8bf*)(Pw + lh * KC + 32 + koff);
    pa1.h[1] = *(const v8bf*)(Pw + lh * KC + 40 + koff);

    // ---- O(16x64) += P(16x64) . V(64x64) ----
#pragma unroll
    for (int n = 0; n < 4; ++n) {
      const __bf16* vp = Vl + (n * 16 + lh) * KC + (hi ? 16 : 0);
      o[n] = WMMA_BF16(pa0.v, *(const v16bf*)vp,        o[n]);
      o[n] = WMMA_BF16(pa1.v, *(const v16bf*)(vp + 32), o[n]);
    }

    __syncthreads();                                   // buffer reuse guard
  }

  // ---- epilogue: normalize, write ctx as bf16 [B][S][H*DK] ----
  const int b = bh >> 4, h = bh & 15;
#pragma unroll
  for (int r = 0; r < 8; ++r) {
    float inv = 1.0f / lrow[r];
    __bf16* cp = ctx + (b * SEQ + q0 + r + hi8) * DMODEL + h * DK;
#pragma unroll
    for (int n = 0; n < 4; ++n)
      cp[n * 16 + lh] = (__bf16)(o[n][r] * inv);
  }
}

// ---------------- output projection: out = ctx . W^T ----------------
// grid: (4096/128) * (1024/64) = 512 blocks, 256 threads
__global__ void __launch_bounds__(256)
proj_kernel(const __bf16* __restrict__ ctx,           // [4096][1024] bf16
            const __bf16* __restrict__ Wb,            // [1024][1024] bf16
            float* __restrict__ out) {                // [4096][1024] f32
  __shared__ __align__(16) char lds[2 * 8192];        // W ping-pong [64 j][64 k]
  const unsigned ldsBase = (unsigned)(uintptr_t)(&lds[0]);

  const int rowBlk = blockIdx.x & 31;
  const int colBlk = blockIdx.x >> 5;
  const int t    = threadIdx.x;
  const int wave = t >> 5;
  const int lane = t & 31;
  const int lh   = lane & 15;
  const int hi   = lane >> 4;
  const int koff = hi ? 16 : 0;
  const int hi8  = hi ? 8 : 0;
  const int q0 = rowBlk * 128 + wave * 16;
  const int j0 = colBlk * 64;

  v8f acc[4] = {{}, {}, {}, {}};
  const __bf16* xrow = ctx + (q0 + lh) * DMODEL;
  const int wj = t >> 3, wpart = t & 7;                // pieces t, t+256

  auto stage = [&](int buf, int c) {
    const int k0 = c * KC;
    const unsigned wd = ldsBase + buf * 8192 + t * 16;
    const __bf16* wsrc = Wb + (j0 + wj) * DMODEL + k0 + wpart * 8;
    ASYNC_LDS_B128(wd,         wsrc);
    ASYNC_LDS_B128(wd + 4096u, wsrc + 32 * DMODEL);
  };

  stage(0, 0);
  const int NCH = DMODEL / KC;                         // 16

  for (int i = 0; i < NCH; ++i) {
    const int cur = i & 1;
    if (i + 1 < NCH) {
      stage(cur ^ 1, i + 1);
      asm volatile("s_wait_asynccnt 0x2" ::: "memory");
    } else {
      asm volatile("s_wait_asynccnt 0x0" ::: "memory");
    }
    __syncthreads();

    const __bf16* Wl = (const __bf16*)(lds + cur * 8192);   // [j][64 k]
    const int k0 = i * KC;
    ABHalves a0, a1;
    a0.h[0] = *(const v8bf*)(xrow + k0 + koff);
    a0.h[1] = *(const v8bf*)(xrow + k0 + koff + 8);
    a1.h[0] = *(const v8bf*)(xrow + k0 + 32 + koff);
    a1.h[1] = *(const v8bf*)(xrow + k0 + 40 + koff);
#pragma unroll
    for (int n = 0; n < 4; ++n) {
      const __bf16* wp = Wl + (n * 16 + lh) * KC + (hi ? 16 : 0);
      acc[n] = WMMA_BF16(a0.v, *(const v16bf*)wp,        acc[n]);
      acc[n] = WMMA_BF16(a1.v, *(const v16bf*)(wp + 32), acc[n]);
    }
    __syncthreads();
  }

#pragma unroll
  for (int r = 0; r < 8; ++r) {
#pragma unroll
    for (int n = 0; n < 4; ++n)
      out[(q0 + r + hi8) * DMODEL + j0 + n * 16 + lh] = acc[n][r];
  }
}

// ---------------- launcher ----------------
extern "C" void kernel_launch(void* const* d_in, const int* in_sizes, int n_in,
                              void* d_out, int out_size, void* d_ws, size_t ws_size,
                              hipStream_t stream) {
  const float* q = (const float*)d_in[0];
  const float* k = (const float*)d_in[1];
  const float* v = (const float*)d_in[2];
  const float* w = (const float*)d_in[3];
  float* out = (float*)d_out;

  char* ws = (char*)d_ws;
  const size_t SZ_QKV = (size_t)BATCH * NHEAD * SEQ * DK * sizeof(__bf16); // 8 MB each
  __bf16* Qb  = (__bf16*)(ws);
  __bf16* Kb  = (__bf16*)(ws + SZ_QKV);
  __bf16* Vt  = (__bf16*)(ws + 2 * SZ_QKV);
  __bf16* Wb  = (__bf16*)(ws + 3 * SZ_QKV);
  __bf16* ctx = (__bf16*)(ws + 3 * SZ_QKV + (size_t)DMODEL * DMODEL * sizeof(__bf16));

  {
    int n = BATCH * SEQ * DMODEL;
    reshape_qkv_kernel<<<(n + 255) / 256, 256, 0, stream>>>(q, k, v, Qb, Kb, Vt);
  }
  {
    int n = DMODEL * DMODEL;
    convert_w_kernel<<<(n + 255) / 256, 256, 0, stream>>>(w, Wb);
  }
  flash_attn_kernel<<<BATCH * NHEAD * (SEQ / 128), 256, 0, stream>>>(Qb, Kb, Vt, ctx);
  proj_kernel<<<(BATCH * SEQ / 128) * (DMODEL / 64), 256, 0, stream>>>(ctx, Wb, out);
}
